// MSResAttnRefineV2_39290360823981
// MI455X (gfx1250) — matrix operand order repaired and checked by reading, not compiled, not run on hardware
//
#include <hip/hip_runtime.h>
#include <cstddef>

typedef __attribute__((ext_vector_type(16))) _Float16 v16h;
typedef __attribute__((ext_vector_type(8)))  float    v8f;

#define TPB 256

// ---------------- weight repack: OIHW f32 -> [K][Cout] f16 (K = cin*9) -------
__global__ void k_w2h(const float* __restrict__ w, _Float16* __restrict__ wk,
                      int K, int Cout, int n) {
  int i = blockIdx.x * blockDim.x + threadIdx.x;
  if (i >= n) return;
  int k = i / Cout, co = i - k * Cout;
  wk[i] = (_Float16)w[(size_t)co * K + k];
}

// ---------------- implicit-GEMM 3x3 conv via WMMA f16 -----------------------
// Block: 256 threads = 8 waves -> 128 pixels x 64 out-channels per block.
// Each wave: 2x2 accumulators (32px x 32co), 4 WMMAs per K-step of 32 with
// A-fragment reuse across co-tiles and B-fragment reuse across px-tiles.
__global__ __launch_bounds__(256) void k_conv_wmma(
    const float* __restrict__ x, const _Float16* __restrict__ wk,
    const float* __restrict__ bias, float* __restrict__ y,
    int Cin, int Hin, int Win, int Cout, int Hout, int Wout, int stride, int act)
{
  __shared__ _Float16 As[128][34];  // 128 px x 32 K (+pad)
  __shared__ _Float16 Bs[32][68];   // 32 K  x 64 co (+pad)
  const int tid   = threadIdx.x;
  const int lane  = tid & 31, wv = tid >> 5;
  const int pw = wv & 3;            // 4 pixel tiles of 32
  const int cw = wv >> 2;           // 2 channel tiles of 32
  const int Npix  = Hout * Wout;
  const int pbase = blockIdx.x * 128;
  const int cobase = blockIdx.y * 64;
  const int K = Cin * 9;
  const int m = lane & 15, hl = lane >> 4;   // wave32 lower/upper half
  v8f zero = {0.f,0.f,0.f,0.f,0.f,0.f,0.f,0.f};
  v8f acc[2][2] = {{zero, zero}, {zero, zero}};

  for (int kt = 0; kt < K; kt += 32) {
    // ---- A tile: im2col gather (pad=1), f32 -> f16. K-column fixed per thread.
    {
      int kk = tid & 31;
      int k = kt + kk;
      bool kv = (k < K);
      int ci = 0, ky = 0, kx = 0;
      if (kv) { ci = k / 9; int r = k - ci * 9; ky = r / 3; kx = r - ky * 3; }
      const float* xc = x + (size_t)ci * Hin * Win;
      for (int pl = tid >> 5; pl < 128; pl += 8) {
        float v = 0.f;
        int p = pbase + pl;
        if (kv && p < Npix) {
          int oy = p / Wout, ox = p - oy * Wout;
          int iy = oy * stride + ky - 1, ix = ox * stride + kx - 1;
          if (iy >= 0 && iy < Hin && ix >= 0 && ix < Win)
            v = xc[(size_t)iy * Win + ix];
        }
        As[pl][kk] = (_Float16)v;
      }
    }
    // ---- B tile: pre-packed [K][Cout] f16, channel fixed per thread.
    {
      int co = tid & 63;
      int c = cobase + co;
      bool cv = (c < Cout);
      for (int kk = tid >> 6; kk < 32; kk += 4) {
        int k = kt + kk;
        _Float16 v = (_Float16)0.f;
        if (cv && k < K) v = wk[(size_t)k * Cout + c];
        Bs[kk][co] = v;
      }
    }
    __syncthreads();

    // ---- fragments in contiguous k order (k = 16*half + s), same for A and B
    v16h fa[2], fb[2];
#pragma unroll
    for (int s = 0; s < 2; ++s) {
#pragma unroll
      for (int v = 0; v < 8; ++v) {
        int ka = hl * 16 + v * 2;
        fa[s][2 * v]     = As[pw * 32 + s * 16 + m][ka];
        fa[s][2 * v + 1] = As[pw * 32 + s * 16 + m][ka + 1];
        int kb = hl * 16 + 2 * v;
        fb[s][2 * v]     = Bs[kb][cw * 32 + s * 16 + m];
        fb[s][2 * v + 1] = Bs[kb + 1][cw * 32 + s * 16 + m];
      }
    }
#pragma unroll
    for (int a = 0; a < 2; ++a)
#pragma unroll
      for (int b = 0; b < 2; ++b)
        acc[a][b] = __builtin_amdgcn_wmma_f32_16x16x32_f16(
            false, fa[a], false, fb[b], (short)0, acc[a][b], false, false);
    __syncthreads();
  }

  // ---- epilogue; C/D layout: VGPR r -> M = r + 8*half, N = lane%16
#pragma unroll
  for (int a = 0; a < 2; ++a) {
#pragma unroll
    for (int b = 0; b < 2; ++b) {
      int c = cobase + cw * 32 + b * 16 + m;
#pragma unroll
      for (int r = 0; r < 8; ++r) {
        int p = pbase + pw * 32 + a * 16 + r + 8 * hl;
        if (p < Npix && c < Cout) {
          float v = acc[a][b][r] + bias[c];
          if (act) v = v > 0.f ? v : 0.2f * v;
          y[(size_t)c * Npix + p] = v;
        }
      }
    }
  }
}

// ---------------- transposed conv 4x4 stride2 (scalar; tiny share of FLOPs) --
__global__ void k_deconv(const float* __restrict__ x, const float* __restrict__ w,
                         const float* __restrict__ bias, float* __restrict__ y,
                         int Cin, int Hin, int Win, int Cout, int act) {
  int Hout = 2 * Hin, Wout = 2 * Win;
  size_t idx = (size_t)blockIdx.x * blockDim.x + threadIdx.x;
  size_t total = (size_t)Cout * Hout * Wout;
  if (idx >= total) return;
  int ox = idx % Wout; size_t t = idx / Wout;
  int oy = t % Hout;   int co = (int)(t / Hout);
  float acc = bias[co];
  for (int ky = 0; ky < 4; ++ky) {
    int d = oy + ky - 2;
    if (d < 0 || (d & 1)) continue;
    int iy = d >> 1; if (iy >= Hin) continue;
    for (int kx = 0; kx < 4; ++kx) {
      int e = ox + kx - 2;
      if (e < 0 || (e & 1)) continue;
      int ix = e >> 1; if (ix >= Win) continue;
      for (int ci = 0; ci < Cin; ++ci) {
        float wv = w[(((size_t)ci * Cout + co) * 4 + (3 - ky)) * 4 + (3 - kx)];
        acc += wv * x[((size_t)ci * Hin + iy) * Win + ix];
      }
    }
  }
  if (act) acc = acc > 0.f ? acc : 0.2f * acc;
  y[idx] = acc;
}

// ---------------- elementwise / glue ----------------------------------------
__global__ void k_copy(const float* __restrict__ s, float* __restrict__ d, size_t n) {
  size_t i = (size_t)blockIdx.x * blockDim.x + threadIdx.x;
  if (i < n) d[i] = s[i];
}
__global__ void k_add(const float* __restrict__ a, const float* __restrict__ b,
                      float* __restrict__ o, size_t n) {
  size_t i = (size_t)blockIdx.x * blockDim.x + threadIdx.x;
  if (i < n) o[i] = a[i] + b[i];
}
__global__ void k_norm(const float* __restrict__ x, float* __restrict__ y,
                       int C, int HW) {
  int p = blockIdx.x * blockDim.x + threadIdx.x;
  if (p >= HW) return;
  float s = 0.f;
  for (int c = 0; c < C; ++c) { float v = x[(size_t)c * HW + p]; s += v * v; }
  float inv = rsqrtf(s);
  for (int c = 0; c < C; ++c) y[(size_t)c * HW + p] = x[(size_t)c * HW + p] * inv;
}
// sim[h][w][k50]: k50<25 -> f-patches, else b-patches (5x5, pad 2)
__global__ void k_corr(const float* __restrict__ xn, const float* __restrict__ fn,
                       const float* __restrict__ bn, float* __restrict__ sim,
                       int C, int H, int W) {
  int idx = blockIdx.x * blockDim.x + threadIdx.x;
  int total = H * W * 50;
  if (idx >= total) return;
  int k50 = idx % 50; int p = idx / 50;
  int w = p % W, h = p / W;
  int which = k50 / 25, k = k50 % 25;
  int dy = k / 5 - 2, dx = k % 5 - 2;
  int sy = h + dy, sx = w + dx;
  float s = 0.f;
  if (sy >= 0 && sy < H && sx >= 0 && sx < W) {
    const float* t = which ? bn : fn;
    for (int c = 0; c < C; ++c)
      s += t[((size_t)c * H + sy) * W + sx] * xn[((size_t)c * H + h) * W + w];
  }
  sim[idx] = s;
}
// align-corners bilinear upsample on [H][W][K] layout
__global__ void k_upsample(const float* __restrict__ in, float* __restrict__ out,
                           int Hin, int Win, int K, int scale) {
  int Hout = Hin * scale, Wout = Win * scale;
  int idx = blockIdx.x * blockDim.x + threadIdx.x;
  int total = Hout * Wout * K;
  if (idx >= total) return;
  int k = idx % K; int p = idx / K;
  int wo = p % Wout, ho = p / Wout;
  float ry = (Hout > 1) ? (float)(Hin - 1) / (float)(Hout - 1) : 0.f;
  float rx = (Wout > 1) ? (float)(Win - 1) / (float)(Wout - 1) : 0.f;
  float sy = ho * ry, sx = wo * rx;
  int y0 = (int)floorf(sy); if (y0 > Hin - 2) y0 = Hin - 2; if (y0 < 0) y0 = 0;
  int x0 = (int)floorf(sx); if (x0 > Win - 2) x0 = Win - 2; if (x0 < 0) x0 = 0;
  float fy = sy - y0, fx = sx - x0;
  const float* q = in;
  float v00 = q[((size_t)y0 * Win + x0) * K + k];
  float v01 = q[((size_t)y0 * Win + x0 + 1) * K + k];
  float v10 = q[((size_t)(y0 + 1) * Win + x0) * K + k];
  float v11 = q[((size_t)(y0 + 1) * Win + x0 + 1) * K + k];
  out[idx] = (1.f - fy) * ((1.f - fx) * v00 + fx * v01) +
             fy * ((1.f - fx) * v10 + fx * v11);
}
__global__ void k_softmax50(float* __restrict__ p, int npix) {
  int i = blockIdx.x * blockDim.x + threadIdx.x;
  if (i >= npix) return;
  float* q = p + (size_t)i * 50;
  float mx = q[0];
  for (int k = 1; k < 50; ++k) mx = fmaxf(mx, q[k]);
  float s = 0.f;
  for (int k = 0; k < 50; ++k) { float e = expf(q[k] - mx); q[k] = e; s += e; }
  float inv = 1.f / s;
  for (int k = 0; k < 50; ++k) q[k] *= inv;
}
// wgt[c,h,w] = sum_k patches5(ff)*pf + patches5(bf)*pb
__global__ void k_wprob(const float* __restrict__ ff, const float* __restrict__ bf,
                        const float* __restrict__ prob, float* __restrict__ out,
                        int C, int H, int W) {
  size_t idx = (size_t)blockIdx.x * blockDim.x + threadIdx.x;
  size_t total = (size_t)C * H * W;
  if (idx >= total) return;
  int p = (int)(idx % (size_t)(H * W)); int c = (int)(idx / (size_t)(H * W));
  int w = p % W, h = p / W;
  const float* pr = prob + (size_t)p * 50;
  float acc = 0.f;
  for (int k = 0; k < 25; ++k) {
    int dy = k / 5 - 2, dx = k % 5 - 2;
    int sy = h + dy, sx = w + dx;
    if (sy >= 0 && sy < H && sx >= 0 && sx < W) {
      size_t o = ((size_t)c * H + sy) * W + sx;
      acc += ff[o] * pr[k] + bf[o] * pr[25 + k];
    }
  }
  out[idx] = acc;
}
// weight_low: normalized patch-weighted ff/bf (denominator sums all 25 probs)
__global__ void k_wlow(const float* __restrict__ ff, const float* __restrict__ bf,
                       const float* __restrict__ low, float* __restrict__ fo,
                       float* __restrict__ bo, int C, int H, int W) {
  size_t idx = (size_t)blockIdx.x * blockDim.x + threadIdx.x;
  size_t total = (size_t)C * H * W;
  if (idx >= total) return;
  int p = (int)(idx % (size_t)(H * W)); int c = (int)(idx / (size_t)(H * W));
  int w = p % W, h = p / W;
  const float* pr = low + (size_t)p * 50;
  float af = 0.f, ab = 0.f, sf = 0.f, sb = 0.f;
  for (int k = 0; k < 25; ++k) {
    float pf = pr[k], pb = pr[25 + k];
    sf += pf; sb += pb;
    int dy = k / 5 - 2, dx = k % 5 - 2;
    int sy = h + dy, sx = w + dx;
    if (sy >= 0 && sy < H && sx >= 0 && sx < W) {
      size_t o = ((size_t)c * H + sy) * W + sx;
      af += ff[o] * pf; ab += bf[o] * pb;
    }
  }
  fo[idx] = af / sf;
  bo[idx] = ab / sb;
}

// ---------------- host orchestration ----------------------------------------
struct Tn { float* p; int C, H, W; };

static inline dim3 gs(size_t n) { return dim3((unsigned)((n + TPB - 1) / TPB)); }

extern "C" void kernel_launch(void* const* d_in, const int* in_sizes, int n_in,
                              void* d_out, int out_size, void* d_ws, size_t ws_size,
                              hipStream_t stream) {
  (void)in_sizes; (void)n_in; (void)out_size; (void)ws_size;
  const int H0 = 192, W0 = 320;
  const size_t HW0 = (size_t)H0 * W0;

  // flattened layer table (BLOCKS order): kind(0=conv,1=deconv), cin,cout,stride,act
  static const int LK[38] = {0,0, 0,0, 0,0,0, 0,0,0, 0,0,0, 0,0,0, 0,0,0,
                             0,0, 0,0, 0,0, 0,0, 0,0, 1,0,0, 1,0,0, 0,0,0};
  static const int LCI[38] = {23,32, 32,32, 32,64,64, 64,128,128, 32,64,128,
                              64,128,256, 128,256,512, 32,32, 64,64, 64,64,
                              128,128, 256,256, 128,64,64, 64,32,32, 32,32,32};
  static const int LCO[38] = {32,32, 32,32, 64,64,64, 128,128,128, 64,128,128,
                              128,256,256, 256,512,512, 32,32, 64,64, 64,32,
                              128,64, 256,128, 64,64,64, 32,32,32, 32,32,3};
  static const int LS[38]  = {1,1, 1,1, 2,1,1, 2,1,1, 2,2,1, 2,2,1, 2,2,1,
                              1,1, 1,1, 1,1, 1,1, 1,1, 2,1,1, 2,1,1, 1,1,1};
  static const int LA[38]  = {1,1, 1,1, 1,1,1, 1,1,1, 1,1,1, 1,1,1, 1,1,1,
                              1,1, 1,1, 1,1, 1,1, 1,1, 1,1,1, 1,1,1, 1,1,0};

  char* base = (char*)d_ws;
  size_t off = 0;
  auto allocB = [&](size_t bytes) -> char* {
    off = (off + 255) & ~(size_t)255;
    char* p = base + off; off += bytes; return p;
  };
  auto allocF = [&](size_t n) -> float* { return (float*)allocB(n * sizeof(float)); };

  // repack all conv weights to f16 [K][Cout]
  _Float16* wkp[38];
  for (int j = 0; j < 38; ++j) {
    wkp[j] = nullptr;
    if (LK[j] == 0) {
      int K = LCI[j] * 9, Cout = LCO[j], n = K * Cout;
      wkp[j] = (_Float16*)allocB((size_t)n * sizeof(_Float16));
      k_w2h<<<gs(n), TPB, 0, stream>>>((const float*)d_in[4 + 2 * j], wkp[j], K, Cout, n);
    }
  }

  auto convL = [&](int j, Tn in) -> Tn {
    int Cout = LCO[j], s = LS[j];
    int Ho = (in.H - 1) / s + 1, Wo = (in.W - 1) / s + 1;
    Tn out{allocF((size_t)Cout * Ho * Wo), Cout, Ho, Wo};
    int Npix = Ho * Wo;
    dim3 g((Npix + 127) / 128, (Cout + 63) / 64);
    k_conv_wmma<<<g, dim3(TPB), 0, stream>>>(
        in.p, wkp[j], (const float*)d_in[5 + 2 * j], out.p,
        in.C, in.H, in.W, Cout, Ho, Wo, s, LA[j]);
    return out;
  };
  auto deconvL = [&](int j, Tn in) -> Tn {
    int Cout = LCO[j];
    int Ho = in.H * 2, Wo = in.W * 2;
    Tn out{allocF((size_t)Cout * Ho * Wo), Cout, Ho, Wo};
    size_t total = (size_t)Cout * Ho * Wo;
    k_deconv<<<gs(total), TPB, 0, stream>>>(
        in.p, (const float*)d_in[4 + 2 * j], (const float*)d_in[5 + 2 * j],
        out.p, in.C, in.H, in.W, Cout, LA[j]);
    return out;
  };
  auto runBlock = [&](int bstart, int blen, Tn in) -> Tn {
    Tn cur = in;
    for (int t = 0; t < blen; ++t) {
      int j = bstart + t;
      cur = (LK[j] == 0) ? convL(j, cur) : deconvL(j, cur);
    }
    return cur;
  };

  // --- encode three streams ---
  auto enc = [&](const float* img, const float* seg, Tn lv[3]) {
    Tn in{allocF(23 * HW0), 23, H0, W0};
    k_copy<<<gs(3 * HW0), TPB, 0, stream>>>(img, in.p, 3 * HW0);
    k_copy<<<gs(20 * HW0), TPB, 0, stream>>>(seg, in.p + 3 * HW0, 20 * HW0);
    Tn a  = runBlock(0, 2, in);
    Tn e1 = runBlock(2, 2, a);
    Tn e2 = runBlock(4, 3, e1);
    Tn e3 = runBlock(7, 3, e2);
    lv[0] = e3; lv[1] = e2; lv[2] = e1;
  };

  const float* cimg = (const float*)d_in[0];
  const float* cseg = (const float*)d_in[1];
  const float* nimg = (const float*)d_in[2];
  const float* nseg = (const float*)d_in[3];

  Tn xf[3], ff[3], bf[3];
  enc(cimg, cseg, xf);
  enc(nimg,           nseg,            ff);
  enc(nimg + 3 * HW0, nseg + 20 * HW0, bf);

  auto corrmap = [&](Tn xd, Tn fd, Tn bd) -> float* {
    int C = xd.C, h = xd.H, w = xd.W, HWp = h * w;
    float* xn = allocF((size_t)C * HWp);
    float* fn = allocF((size_t)C * HWp);
    float* bn = allocF((size_t)C * HWp);
    k_norm<<<gs(HWp), TPB, 0, stream>>>(xd.p, xn, C, HWp);
    k_norm<<<gs(HWp), TPB, 0, stream>>>(fd.p, fn, C, HWp);
    k_norm<<<gs(HWp), TPB, 0, stream>>>(bd.p, bn, C, HWp);
    float* sim = allocF((size_t)HWp * 50);
    k_corr<<<gs((size_t)HWp * 50), TPB, 0, stream>>>(xn, fn, bn, sim, C, h, w);
    int Ho = h * 4, Wo = w * 4;
    float* up = allocF((size_t)Ho * Wo * 50);
    k_upsample<<<gs((size_t)Ho * Wo * 50), TPB, 0, stream>>>(sim, up, h, w, 50, 4);
    k_softmax50<<<gs((size_t)Ho * Wo), TPB, 0, stream>>>(up, Ho * Wo);
    return up;
  };

  float* probs[3]; int probH[3], probW[3]; int nprob = 0;
  Tn fused[3];
  for (int i = 0; i < 3; ++i) {
    int lvl = 3 - i;
    int C = ff[i].C, Hl = ff[i].H, Wl = ff[i].W;
    size_t cel = (size_t)C * Hl * Wl;
    // fold in previous prob maps (weight_low)
    for (int k = 0; k < nprob; ++k) {
      int scale = 1 << (nprob - k);
      int Ho = probH[k] * scale, Wo = probW[k] * scale;
      float* low = allocF((size_t)Ho * Wo * 50);
      k_upsample<<<gs((size_t)Ho * Wo * 50), TPB, 0, stream>>>(
          probs[k], low, probH[k], probW[k], 50, scale);
      float* fo = allocF(cel); float* bo = allocF(cel);
      k_wlow<<<gs(cel), TPB, 0, stream>>>(ff[i].p, bf[i].p, low, fo, bo, C, Hl, Wl);
      ff[i].p = fo; bf[i].p = bo;
    }
    if (i != 0) {
      int nb = (lvl == 1) ? 19 : 21;
      ff[i] = runBlock(nb, 2, ff[i]);
      bf[i] = runBlock(nb, 2, bf[i]);
    }
    int ad = (lvl == 1) ? 10 : (lvl == 2) ? 13 : 16;
    Tn xd = runBlock(ad, 3, xf[i]);
    Tn fd = runBlock(ad, 3, ff[i]);
    Tn bd = runBlock(ad, 3, bf[i]);
    float* prob = corrmap(xd, fd, bd);
    probs[nprob] = prob; probH[nprob] = xd.H * 4; probW[nprob] = xd.W * 4; nprob++;

    Tn wgt{allocF(cel), C, Hl, Wl};
    k_wprob<<<gs(cel), TPB, 0, stream>>>(ff[i].p, bf[i].p, prob, wgt.p, C, Hl, Wl);

    Tn cc{allocF(2 * cel), 2 * C, Hl, Wl};
    k_copy<<<gs(cel), TPB, 0, stream>>>(xf[i].p, cc.p, cel);
    k_copy<<<gs(cel), TPB, 0, stream>>>(wgt.p, cc.p + cel, cel);
    int fu = (lvl == 1) ? 23 : (lvl == 2) ? 25 : 27;
    fused[i] = runBlock(fu, 2, cc);
  }

  // decoder + residual output
  Tn d3 = runBlock(29, 3, fused[0]);                 // 64 @ 96x160
  size_t n2 = (size_t)64 * 96 * 160;
  Tn f2{allocF(n2), 64, 96, 160};
  k_add<<<gs(n2), TPB, 0, stream>>>(fused[1].p, d3.p, f2.p, n2);
  Tn d2 = runBlock(32, 3, f2);                       // 32 @ 192x320
  size_t n1 = (size_t)32 * HW0;
  Tn f1{allocF(n1), 32, H0, W0};
  k_add<<<gs(n1), TPB, 0, stream>>>(fused[2].p, d2.p, f1.p, n1);
  Tn o = runBlock(35, 3, f1);                        // 3 @ 192x320
  k_add<<<gs(3 * HW0), TPB, 0, stream>>>(o.p, cimg, (float*)d_out, 3 * HW0);
}
